// SoftMoEPolicyNetwork_42992622633676
// MI455X (gfx1250) — matrix operand order repaired and checked by reading, not compiled, not run on hardware
//
#include <hip/hip_runtime.h>
#include <hip/hip_bf16.h>
#include <math.h>

// ---------------------------------------------------------------------------
// SoftMoEPolicyNetwork forward on MI455X (gfx1250, wave32, WMMA bf16).
// All GEMMs (M=32768) run on V_WMMA_F32_16X16X32_BF16 with fp32 accumulation.
// Each wave computes a 32x32 output tile (2x2 WMMAs, fragment reuse);
// block = 8 waves -> 128x64 tile. K staged via float4 -> packed bf16 -> b64
// LDS stores; guard-free main K-loop + one guarded tail tile (K=272 case).
// ---------------------------------------------------------------------------

typedef __attribute__((ext_vector_type(16))) __bf16          v16bf;
typedef __attribute__((ext_vector_type(16))) unsigned short  v16us;
typedef __attribute__((ext_vector_type(8)))  float           v8f;

constexpr int   Bn   = 32768;

// ---- workspace layout (floats) --------------------------------------------
constexpr size_t O_H   = 0;                               // [B,256] h, later moe
constexpr size_t O_LOG = O_H   + (size_t)Bn * 256;        // [B,12] router logits
constexpr size_t O_W12 = O_LOG + (size_t)Bn * 12;         // [B,12] sparse gates
constexpr size_t O_T1  = O_W12 + (size_t)Bn * 12;         // [B,128]
constexpr size_t O_T2  = O_T1  + (size_t)Bn * 128;        // [B,64]
constexpr size_t O_EO  = O_T2  + (size_t)Bn * 64;         // [B,64]
constexpr size_t O_F   = O_EO  + (size_t)Bn * 64;         // [B,64] moe accum
constexpr size_t O_TI  = O_F   + (size_t)Bn * 64;         // [B,272] thought/conf in
constexpr size_t O_H1  = O_TI  + (size_t)Bn * 272;        // [B,256] GRU h1
constexpr size_t O_H2  = O_H1  + (size_t)Bn * 256;        // [B,256] GRU h2
constexpr size_t O_G0  = O_H2  + (size_t)Bn * 256;        // [B,768] gi (layer1, const)
constexpr size_t O_G1  = O_G0  + (size_t)Bn * 768;        // [B,768]
constexpr size_t O_G2  = O_G1  + (size_t)Bn * 768;        // [B,768]
constexpr size_t O_HD  = O_G2  + (size_t)Bn * 768;        // [B,64] head hidden
constexpr size_t O_MT  = O_HD  + (size_t)Bn * 64;         // [B,2] meta
constexpr size_t O_BAL = O_MT  + (size_t)Bn * 2;          // [12] balance accum

// ---- output layout (floats, concat of reference tuple) --------------------
constexpr size_t Q_FIN = 0;                                // [B,16]
constexpr size_t Q_VER = Q_FIN + (size_t)Bn * 16;          // [B,16]
constexpr size_t Q_S1  = Q_VER + (size_t)Bn * 16;          // [B,16]
constexpr size_t Q_VAL = Q_S1  + (size_t)Bn * 16;          // [B,1]
constexpr size_t Q_PR  = Q_VAL + (size_t)Bn;               // [B,1]
constexpr size_t Q_CF  = Q_PR  + (size_t)Bn;               // [B,1]
constexpr size_t Q_BAL = Q_CF  + (size_t)Bn;               // [1]
constexpr size_t Q_S1C = Q_BAL + 1;                        // [B]
constexpr size_t Q_S2C = Q_S1C + (size_t)Bn;               // [B]

// ---- helpers --------------------------------------------------------------
__device__ __forceinline__ unsigned int f2bf(float f) {
  unsigned int u = __builtin_bit_cast(unsigned int, f);
  unsigned int r = u + 0x7FFFu + ((u >> 16) & 1u);          // round-to-nearest-even
  return r >> 16;
}
__device__ __forceinline__ float gelu_exact(float x) {
  return 0.5f * x * (1.0f + erff(x * 0.70710678118654752440f));
}
__device__ __forceinline__ float sigmoidf(float x) {
  return 1.0f / (1.0f + expf(-x));
}

// ---------------------------------------------------------------------------
// WMMA GEMM: C[M,N] = act(A[M,K] * W[N,K]^T + bias), fp32 in/out, bf16 MACs.
// Block tile 128x64, wave tile 32x32 (2x2 x 16x16 WMMA). K tile = 32.
// ---------------------------------------------------------------------------
#define BM 128
#define BN 64
#define BK 32
#define APAD 36   // row pitch in ushorts (32 data + 4 pad, keeps 8B alignment)

template <bool GUARD>
__device__ __forceinline__ void stage_tiles(
    const float* __restrict__ A, const float* __restrict__ W,
    unsigned short (*sA)[APAD], unsigned short (*sB)[APAD],
    int tid, int m0, int n0, int k0, int K, int lda)
{
  // A tile: 128x32 floats = 1024 float4, 4 per thread (coalesced 8xfloat4/row)
#pragma unroll
  for (int i = 0; i < 4; ++i) {
    int q  = tid + i * 256;
    int r  = q >> 3;
    int c4 = (q & 7) * 4;
    int gc = k0 + c4;
    float4 f = make_float4(0.f, 0.f, 0.f, 0.f);
    if (!GUARD || gc < K)                 // K % 4 == 0 for every GEMM here
      f = *(const float4*)(A + (size_t)(m0 + r) * lda + gc);
    unsigned int lo = f2bf(f.x) | (f2bf(f.y) << 16);
    unsigned int hi = f2bf(f.z) | (f2bf(f.w) << 16);
    *(uint2*)&sA[r][c4] = make_uint2(lo, hi);   // single ds_store_b64
  }
  // W tile: 64x32 floats = 512 float4, 2 per thread
#pragma unroll
  for (int i = 0; i < 2; ++i) {
    int q  = tid + i * 256;
    int r  = q >> 3;
    int c4 = (q & 7) * 4;
    int gc = k0 + c4;
    float4 f = make_float4(0.f, 0.f, 0.f, 0.f);
    if (!GUARD || gc < K)
      f = *(const float4*)(W + (size_t)(n0 + r) * K + gc);
    unsigned int lo = f2bf(f.x) | (f2bf(f.y) << 16);
    unsigned int hi = f2bf(f.z) | (f2bf(f.w) << 16);
    *(uint2*)&sB[r][c4] = make_uint2(lo, hi);
  }
}

__device__ __forceinline__ void wmma_step(
    unsigned short (*sA)[APAD], unsigned short (*sB)[APAD],
    int wm, int wn, int half, int l16, v8f acc[2][2])
{
  v16us au[2], bu[2];
#pragma unroll
  for (int t = 0; t < 2; ++t) {
#pragma unroll
    for (int e = 0; e < 16; ++e) {
      int v = e >> 1, s = e & 1;
      int ka = ((v & 3) * 2) + ((v >> 2) * 16) + half * 8 + s;  // A frag K idx
      int kb = ((v & 3) * 2) + ((v >> 2) * 16) + half * 8 + s;  // B frag K idx
      au[t][e] = sA[wm * 32 + t * 16 + l16][ka];
      bu[t][e] = sB[wn * 32 + t * 16 + l16][kb];
    }
  }
#pragma unroll
  for (int mi = 0; mi < 2; ++mi)
#pragma unroll
    for (int ni = 0; ni < 2; ++ni)
      acc[mi][ni] = __builtin_amdgcn_wmma_f32_16x16x32_bf16(
          false, __builtin_bit_cast(v16bf, au[mi]),
          false, __builtin_bit_cast(v16bf, bu[ni]),
          (short)0, acc[mi][ni], false, false);
}

__global__ __launch_bounds__(256)
void gemm_bf16_kernel(const float* __restrict__ A, const float* __restrict__ W,
                      const float* __restrict__ bias, float* __restrict__ C,
                      int N, int K, int lda, int ldc, int act)
{
  __shared__ unsigned short sA[BM][APAD];
  __shared__ unsigned short sB[BN][APAD];

  const int tid  = threadIdx.x;
  const int lane = tid & 31;
  const int wave = tid >> 5;
  const int wm   = wave & 3;        // 0..3 : 32-row slice within 128-row tile
  const int wn   = wave >> 2;       // 0..1 : 32-col slice within 64-col tile
  const int half = lane >> 4;
  const int l16  = lane & 15;
  const int m0   = blockIdx.y * BM;
  const int n0   = blockIdx.x * BN;

  v8f acc[2][2] = {};

  const int K32 = K & ~(BK - 1);
  for (int k0 = 0; k0 < K32; k0 += BK) {
    stage_tiles<false>(A, W, sA, sB, tid, m0, n0, k0, K, lda);
    __syncthreads();
    wmma_step(sA, sB, wm, wn, half, l16, acc);
    __syncthreads();
  }
  if (K & (BK - 1)) {                       // zero-padded tail (e.g. K=272)
    stage_tiles<true>(A, W, sA, sB, tid, m0, n0, K32, K, lda);
    __syncthreads();
    wmma_step(sA, sB, wm, wn, half, l16, acc);
  }

  // epilogue: bias + optional activation; C layout per ISA 7.12.2
#pragma unroll
  for (int mi = 0; mi < 2; ++mi) {
#pragma unroll
    for (int ni = 0; ni < 2; ++ni) {
      int gn = n0 + wn * 32 + ni * 16 + l16;
      float bv = bias ? bias[gn] : 0.0f;
#pragma unroll
      for (int r = 0; r < 8; ++r) {
        int gm = m0 + wm * 32 + mi * 16 + half * 8 + r;
        float v = acc[mi][ni][r] + bv;
        if (act == 1) v = gelu_exact(v);
        else if (act == 2) v = sigmoidf(v);
        C[(size_t)gm * ldc + gn] = v;
      }
    }
  }
}

// ---------------------------------------------------------------------------
// LayerNorm (+ optional exact GELU), in place. One wave per row (D<=256).
// ---------------------------------------------------------------------------
__global__ __launch_bounds__(256)
void ln_act_kernel(float* __restrict__ X, const float* __restrict__ g,
                   const float* __restrict__ be, int D, int rows, int act)
{
  int row  = blockIdx.x * 8 + (threadIdx.x >> 5);
  int lane = threadIdx.x & 31;
  if (row >= rows) return;
  float* p = X + (size_t)row * D;
  float s = 0.0f, s2 = 0.0f;
  for (int j = lane; j < D; j += 32) { float v = p[j]; s += v; s2 += v * v; }
#pragma unroll
  for (int o = 16; o > 0; o >>= 1) {
    s  += __shfl_xor(s,  o, 32);
    s2 += __shfl_xor(s2, o, 32);
  }
  float mean = s / D;
  float var  = s2 / D - mean * mean;
  float inv  = rsqrtf(var + 1e-5f);
  for (int j = lane; j < D; j += 32) {
    float v = (p[j] - mean) * inv * g[j] + be[j];
    if (act) v = gelu_exact(v);
    p[j] = v;
  }
}

// ---------------------------------------------------------------------------
// Router: full softmax (balance-loss mass via LDS + global atomics),
// top-2 + softmax gates -> sparse weight rows w[B,12].
// ---------------------------------------------------------------------------
__global__ __launch_bounds__(256)
void router_kernel(const float* __restrict__ logits, float* __restrict__ w,
                   float* __restrict__ bal, int B)
{
  __shared__ float sb[12];
  if (threadIdx.x < 12) sb[threadIdx.x] = 0.0f;
  __syncthreads();
  int b = blockIdx.x * 256 + threadIdx.x;
  if (b < B) {
    float l[12], p[12];
    float mx = -1e30f;
#pragma unroll
    for (int e = 0; e < 12; ++e) { l[e] = logits[(size_t)b * 12 + e]; mx = fmaxf(mx, l[e]); }
    float sum = 0.0f;
#pragma unroll
    for (int e = 0; e < 12; ++e) { p[e] = expf(l[e] - mx); sum += p[e]; }
    float inv = 1.0f / sum;
#pragma unroll
    for (int e = 0; e < 12; ++e) atomicAdd(&sb[e], p[e] * inv);
    int i0 = 0;
#pragma unroll
    for (int e = 1; e < 12; ++e) if (l[e] > l[i0]) i0 = e;
    int i1 = (i0 == 0) ? 1 : 0;
#pragma unroll
    for (int e = 0; e < 12; ++e) { if (e != i0 && l[e] > l[i1]) i1 = e; }
    float m2 = fmaxf(l[i0], l[i1]);
    float e0 = expf(l[i0] - m2), e1 = expf(l[i1] - m2);
    float n2 = 1.0f / (e0 + e1);
#pragma unroll
    for (int e = 0; e < 12; ++e) w[(size_t)b * 12 + e] = 0.0f;
    w[(size_t)b * 12 + i0] = e0 * n2;
    w[(size_t)b * 12 + i1] = e1 * n2;
  }
  __syncthreads();
  if (threadIdx.x < 12) atomicAdd(&bal[threadIdx.x], sb[threadIdx.x]);
}

// moeF[b,:] += w[b,e] * eo[b,:]
__global__ void axpy_gate_kernel(float* __restrict__ F, const float* __restrict__ eo,
                                 const float* __restrict__ w, int e, int B)
{
  int idx = blockIdx.x * 256 + threadIdx.x;
  if (idx >= B * 64) return;
  int b = idx >> 6;
  F[idx] += w[(size_t)b * 12 + e] * eo[idx];
}

// tiny dense heads (N <= 16): out[b,n] = act(dot(A[b,:],W[n,:]) + bias[n])
__global__ void small_head_kernel(const float* __restrict__ A, const float* __restrict__ W,
                                  const float* __restrict__ bias, float* __restrict__ out,
                                  int B, int K, int N, int act, int ldo)
{
  int idx = blockIdx.x * 256 + threadIdx.x;
  if (idx >= B * N) return;
  int b = idx / N, n = idx % N;
  const float* a  = A + (size_t)b * K;
  const float* ww = W + (size_t)n * K;
  float s = bias ? bias[n] : 0.0f;
  for (int k = 0; k < K; ++k) s = fmaf(a[k], ww[k], s);
  if (act == 2) s = sigmoidf(s);
  out[(size_t)b * ldo + n] = s;
}

// GRU gate combine: h = (1-z)*n + z*h
__global__ void gru_combine_kernel(const float* __restrict__ gi, const float* __restrict__ gh,
                                   float* __restrict__ h, int B)
{
  int idx = blockIdx.x * 256 + threadIdx.x;
  if (idx >= B * 256) return;
  int b = idx >> 8, j = idx & 255;
  const float* pi = gi + (size_t)b * 768;
  const float* ph = gh + (size_t)b * 768;
  float r  = sigmoidf(pi[j]       + ph[j]);
  float zg = sigmoidf(pi[j + 256] + ph[j + 256]);
  float n  = tanhf(pi[j + 512] + r * ph[j + 512]);
  h[idx] = (1.0f - zg) * n + zg * h[idx];
}

// ti[:, :256] = moe
__global__ void copy_moe_ti_kernel(const float* __restrict__ moe, float* __restrict__ ti, int B)
{
  int idx = blockIdx.x * 256 + threadIdx.x;
  if (idx >= B * 256) return;
  int b = idx >> 8, j = idx & 255;
  ti[(size_t)b * 272 + j] = moe[idx];
}

// softmax(s1_logits) -> ti[:,256:272]; s1_conf = max prob
__global__ void s1_post_kernel(const float* __restrict__ s1, float* __restrict__ ti,
                               float* __restrict__ s1c, int B)
{
  int b = blockIdx.x * 256 + threadIdx.x;
  if (b >= B) return;
  float l[16]; float mx = -1e30f;
#pragma unroll
  for (int i = 0; i < 16; ++i) { l[i] = s1[(size_t)b * 16 + i]; mx = fmaxf(mx, l[i]); }
  float s = 0.0f;
#pragma unroll
  for (int i = 0; i < 16; ++i) { l[i] = expf(l[i] - mx); s += l[i]; }
  float inv = 1.0f / s; float best = 0.0f;
#pragma unroll
  for (int i = 0; i < 16; ++i) {
    float p = l[i] * inv;
    ti[(size_t)b * 272 + 256 + i] = p;
    best = fmaxf(best, p);
  }
  s1c[b] = best;
}

// meta softmax + final combine; s2_conf; final_logits also into ti[:,256:272]
__global__ void meta_final_kernel(const float* __restrict__ meta, const float* __restrict__ s1,
                                  const float* __restrict__ ver, float* __restrict__ fin,
                                  float* __restrict__ s2c, float* __restrict__ ti, int B)
{
  int b = blockIdx.x * 256 + threadIdx.x;
  if (b >= B) return;
  float m0 = meta[(size_t)b * 2], m1 = meta[(size_t)b * 2 + 1];
  float mm = fmaxf(m0, m1);
  float e0 = expf(m0 - mm), e1 = expf(m1 - mm);
  float inv = 1.0f / (e0 + e1);
  m0 = e0 * inv; m1 = e1 * inv;
  float l[16]; float mx = -1e30f;
#pragma unroll
  for (int i = 0; i < 16; ++i) { l[i] = ver[(size_t)b * 16 + i]; mx = fmaxf(mx, l[i]); }
  float s = 0.0f;
#pragma unroll
  for (int i = 0; i < 16; ++i) { l[i] = expf(l[i] - mx); s += l[i]; }
  float best = 0.0f;
#pragma unroll
  for (int i = 0; i < 16; ++i) best = fmaxf(best, l[i] / s);
  s2c[b] = best;
#pragma unroll
  for (int i = 0; i < 16; ++i) {
    float f = m0 * s1[(size_t)b * 16 + i] + m1 * ver[(size_t)b * 16 + i];
    fin[(size_t)b * 16 + i] = f;
    ti[(size_t)b * 272 + 256 + i] = f;   // confidence input = [moe, final_logits]
  }
}

__global__ void zero_kernel(float* __restrict__ p, size_t n)
{
  size_t i = (size_t)blockIdx.x * 256 + threadIdx.x;
  if (i < n) p[i] = 0.0f;
}

__global__ void bal_finalize_kernel(const float* __restrict__ acc, float* __restrict__ out, int B)
{
  if (blockIdx.x == 0 && threadIdx.x == 0) {
    float s = 0.0f;
#pragma unroll
    for (int e = 0; e < 12; ++e) {
      float d = acc[e] / (float)B - (1.0f / 12.0f);
      s += d * d;
    }
    out[0] = s / 12.0f;
  }
}

// ---------------------------------------------------------------------------
extern "C" void kernel_launch(void* const* d_in, const int* in_sizes, int n_in,
                              void* d_out, int out_size, void* d_ws, size_t ws_size,
                              hipStream_t stream)
{
  (void)in_sizes; (void)n_in; (void)out_size; (void)ws_size;
  const float* x       = (const float*)d_in[0];
  const float* proj_W  = (const float*)d_in[1];
  const float* proj_b  = (const float*)d_in[2];
  const float* proj_g  = (const float*)d_in[3];
  const float* proj_be = (const float*)d_in[4];
  const float* rout_W  = (const float*)d_in[5];
  const float* rout_b  = (const float*)d_in[6];
  const float* eW1     = (const float*)d_in[7];
  const float* eb1     = (const float*)d_in[8];
  const float* eg1     = (const float*)d_in[9];
  const float* ebe1    = (const float*)d_in[10];
  const float* eW2     = (const float*)d_in[11];
  const float* eb2     = (const float*)d_in[12];
  const float* eg2     = (const float*)d_in[13];
  const float* ebe2    = (const float*)d_in[14];
  const float* eW3     = (const float*)d_in[15];
  const float* eb3     = (const float*)d_in[16];
  const float* mp_W    = (const float*)d_in[17];
  const float* mp_b    = (const float*)d_in[18];
  const float* mp_g    = (const float*)d_in[19];
  const float* mp_be   = (const float*)d_in[20];
  const float* s1_W1   = (const float*)d_in[21];
  const float* s1_b1   = (const float*)d_in[22];
  const float* s1_g    = (const float*)d_in[23];
  const float* s1_be   = (const float*)d_in[24];
  const float* s1_W2   = (const float*)d_in[25];
  const float* s1_b2   = (const float*)d_in[26];
  const float* g1_Wih  = (const float*)d_in[27];
  const float* g1_Whh  = (const float*)d_in[28];
  const float* g1_bih  = (const float*)d_in[29];
  const float* g1_bhh  = (const float*)d_in[30];
  const float* g2_Wih  = (const float*)d_in[31];
  const float* g2_Whh  = (const float*)d_in[32];
  const float* g2_bih  = (const float*)d_in[33];
  const float* g2_bhh  = (const float*)d_in[34];
  const float* v_W1    = (const float*)d_in[35];
  const float* v_b1    = (const float*)d_in[36];
  const float* v_g     = (const float*)d_in[37];
  const float* v_be    = (const float*)d_in[38];
  const float* v_W2    = (const float*)d_in[39];
  const float* v_b2    = (const float*)d_in[40];
  const float* mw_W1   = (const float*)d_in[41];
  const float* mw_b1   = (const float*)d_in[42];
  const float* mw_W2   = (const float*)d_in[43];
  const float* mw_b2   = (const float*)d_in[44];
  const float* val_W1  = (const float*)d_in[45];
  const float* val_b1  = (const float*)d_in[46];
  const float* val_W2  = (const float*)d_in[47];
  const float* val_b2  = (const float*)d_in[48];
  const float* pr_W1   = (const float*)d_in[49];
  const float* pr_b1   = (const float*)d_in[50];
  const float* pr_W2   = (const float*)d_in[51];
  const float* pr_b2   = (const float*)d_in[52];
  const float* cf_W1   = (const float*)d_in[53];
  const float* cf_b1   = (const float*)d_in[54];
  const float* cf_W2   = (const float*)d_in[55];
  const float* cf_b2   = (const float*)d_in[56];

  float* ws  = (float*)d_ws;
  float* out = (float*)d_out;

  auto gemm = [&](const float* A, const float* W, const float* bias, float* C,
                  int N, int K, int lda, int ldc, int act) {
    dim3 grid(N / BN, Bn / BM);
    gemm_bf16_kernel<<<grid, 256, 0, stream>>>(A, W, bias, C, N, K, lda, ldc, act);
  };
  auto ln = [&](float* X, const float* g, const float* be, int D, int act) {
    ln_act_kernel<<<Bn / 8, 256, 0, stream>>>(X, g, be, D, Bn, act);
  };
  auto zero = [&](float* p, size_t n) {
    zero_kernel<<<(unsigned)((n + 255) / 256), 256, 0, stream>>>(p, n);
  };

  // ---- input projection: Linear(384,256)+LN+GELU --------------------------
  gemm(x, proj_W, proj_b, ws + O_H, 256, 384, 384, 256, 0);
  ln(ws + O_H, proj_g, proj_be, 256, 1);

  // ---- router: Linear(256,12), softmax, top-2 gates, balance loss ---------
  small_head_kernel<<<(Bn * 12 + 255) / 256, 256, 0, stream>>>(
      ws + O_H, rout_W, rout_b, ws + O_LOG, Bn, 256, 12, 0, 12);
  zero(ws + O_BAL, 12);
  router_kernel<<<Bn / 256, 256, 0, stream>>>(ws + O_LOG, ws + O_W12, ws + O_BAL, Bn);
  bal_finalize_kernel<<<1, 32, 0, stream>>>(ws + O_BAL, out + Q_BAL, Bn);

  // ---- 12 experts, gated accumulation into F ------------------------------
  zero(ws + O_F, (size_t)Bn * 64);
  for (int e = 0; e < 12; ++e) {
    gemm(ws + O_H, eW1 + (size_t)e * 128 * 256, eb1 + e * 128, ws + O_T1, 128, 256, 256, 128, 0);
    ln(ws + O_T1, eg1 + e * 128, ebe1 + e * 128, 128, 1);
    gemm(ws + O_T1, eW2 + (size_t)e * 64 * 128, eb2 + e * 64, ws + O_T2, 64, 128, 128, 64, 0);
    ln(ws + O_T2, eg2 + e * 64, ebe2 + e * 64, 64, 1);
    gemm(ws + O_T2, eW3 + (size_t)e * 64 * 64, eb3 + e * 64, ws + O_EO, 64, 64, 64, 64, 0);
    axpy_gate_kernel<<<(Bn * 64) / 256, 256, 0, stream>>>(ws + O_F, ws + O_EO, ws + O_W12, e, Bn);
  }

  // ---- moe output projection: Linear(64,256)+LN+GELU (overwrites h) -------
  gemm(ws + O_F, mp_W, mp_b, ws + O_H, 256, 64, 64, 256, 0);
  ln(ws + O_H, mp_g, mp_be, 256, 1);                      // ws+O_H is now "moe"

  // ---- System 1 -----------------------------------------------------------
  gemm(ws + O_H, s1_W1, s1_b1, ws + O_T1, 128, 256, 256, 128, 0);
  ln(ws + O_T1, s1_g, s1_be, 128, 1);
  small_head_kernel<<<(Bn * 16 + 255) / 256, 256, 0, stream>>>(
      ws + O_T1, s1_W2, s1_b2, out + Q_S1, Bn, 128, 16, 0, 16);

  // ---- thought input ti = [moe, softmax(s1_logits)] -----------------------
  copy_moe_ti_kernel<<<(Bn * 256) / 256, 256, 0, stream>>>(ws + O_H, ws + O_TI, Bn);
  s1_post_kernel<<<Bn / 256, 256, 0, stream>>>(out + Q_S1, ws + O_TI, out + Q_S1C, Bn);

  // ---- System 2: 2-layer GRU, 3 steps; gi(layer1) constant over steps -----
  zero(ws + O_H1, (size_t)Bn * 256);
  zero(ws + O_H2, (size_t)Bn * 256);
  gemm(ws + O_TI, g1_Wih, g1_bih, ws + O_G0, 768, 272, 272, 768, 0);  // K=272 padded
  for (int t = 0; t < 3; ++t) {
    gemm(ws + O_H1, g1_Whh, g1_bhh, ws + O_G1, 768, 256, 256, 768, 0);
    gru_combine_kernel<<<(Bn * 256) / 256, 256, 0, stream>>>(ws + O_G0, ws + O_G1, ws + O_H1, Bn);
    gemm(ws + O_H1, g2_Wih, g2_bih, ws + O_G1, 768, 256, 256, 768, 0);
    gemm(ws + O_H2, g2_Whh, g2_bhh, ws + O_G2, 768, 256, 256, 768, 0);
    gru_combine_kernel<<<(Bn * 256) / 256, 256, 0, stream>>>(ws + O_G1, ws + O_G2, ws + O_H2, Bn);
  }

  // ---- verification head --------------------------------------------------
  gemm(ws + O_H2, v_W1, v_b1, ws + O_T1, 128, 256, 256, 128, 0);
  ln(ws + O_T1, v_g, v_be, 128, 1);
  small_head_kernel<<<(Bn * 16 + 255) / 256, 256, 0, stream>>>(
      ws + O_T1, v_W2, v_b2, out + Q_VER, Bn, 128, 16, 0, 16);

  // ---- meta weights + final combine (writes conf input into ti) -----------
  gemm(ws + O_H, mw_W1, mw_b1, ws + O_HD, 64, 256, 256, 64, 1);   // GELU fused
  small_head_kernel<<<(Bn * 2 + 255) / 256, 256, 0, stream>>>(
      ws + O_HD, mw_W2, mw_b2, ws + O_MT, Bn, 64, 2, 0, 2);
  meta_final_kernel<<<Bn / 256, 256, 0, stream>>>(
      ws + O_MT, out + Q_S1, out + Q_VER, out + Q_FIN, out + Q_S2C, ws + O_TI, Bn);

  // ---- value head ---------------------------------------------------------
  gemm(ws + O_H, val_W1, val_b1, ws + O_HD, 64, 256, 256, 64, 1);
  small_head_kernel<<<(Bn + 255) / 256, 256, 0, stream>>>(
      ws + O_HD, val_W2, val_b2, out + Q_VAL, Bn, 64, 1, 0, 1);

  // ---- process reward head (sigmoid) --------------------------------------
  gemm(ws + O_H, pr_W1, pr_b1, ws + O_HD, 64, 256, 256, 64, 1);
  small_head_kernel<<<(Bn + 255) / 256, 256, 0, stream>>>(
      ws + O_HD, pr_W2, pr_b2, out + Q_PR, Bn, 64, 1, 2, 1);

  // ---- confidence head on [moe, final_logits] (sigmoid) -------------------
  gemm(ws + O_TI, cf_W1, cf_b1, ws + O_HD, 64, 272, 272, 64, 1);   // K=272 padded
  small_head_kernel<<<(Bn + 255) / 256, 256, 0, stream>>>(
      ws + O_HD, cf_W2, cf_b2, out + Q_CF, Bn, 64, 1, 2, 1);
}
